// PerceiverAttention_83399674954359
// MI455X (gfx1250) — matrix operand order repaired and checked by reading, not compiled, not run on hardware
//
#include <hip/hip_runtime.h>
#include <math.h>

#define HID   2048
#define NH    32
#define HD    64
#define NG    8
#define QLEN  64
#define SEQ   4096
#define KVL   4160      // 4096 + 64 == 65*64 (batch boundaries are 64-aligned)
#define BSZ   8
#define KVD   512       // NG*HD
#define TLDA  40        // padded f16 k-stride for A tile (80B: 16B aligned, conflict-free)
#define CLDA  136       // padded f16 stride for C tile (272B = 17*16B: aligned for v8h)

typedef __attribute__((ext_vector_type(16))) _Float16 v16h;
typedef __attribute__((ext_vector_type(8)))  _Float16 v8h;
typedef __attribute__((ext_vector_type(8)))  float    v8f;

union FragH { v16h v; struct { v8h lo; v8h hi; } p; };

// Build a 16-element f16 fragment from two contiguous 16B chunks.
__device__ inline v16h ldfrag(const _Float16* lo, const _Float16* hi) {
  FragH f;
  f.p.lo = *(const v8h*)lo;
  f.p.hi = *(const v8h*)hi;
  return f.v;
}

__device__ inline v8f wmma16(v16h a, v16h b, v8f c) {
  // D = A(16x32 f16) * B(32x16 f16) + C(16x16 f32)
  return __builtin_amdgcn_wmma_f32_16x16x32_f16(false, a, false, b, (short)0, c, false, false);
}

// ---------------------------------------------------------------------------
// Weight transpose+convert: W[2048][N] f32 -> Wt[N][2048] f16, 64x64 LDS tiles.
// Done once; all GEMMs then read f16 B-fragments straight from global (L2-resident).
// ---------------------------------------------------------------------------
__global__ __launch_bounds__(256)
void wtrans_kernel(const float* __restrict__ W, _Float16* __restrict__ Wt, int N)
{
  __shared__ _Float16 T[64][72];
  const int k0 = blockIdx.x * 64;
  const int n0 = blockIdx.y * 64;
  const int t  = threadIdx.x;
  const int r  = t >> 2, seg = (t & 3) * 16;

  {
    const float* src = W + (size_t)(k0 + r) * N + n0 + seg;
    #pragma unroll
    for (int i = 0; i < 16; i += 4) {
      float4 f = *(const float4*)(src + i);
      T[r][seg + i + 0] = (_Float16)f.x;
      T[r][seg + i + 1] = (_Float16)f.y;
      T[r][seg + i + 2] = (_Float16)f.z;
      T[r][seg + i + 3] = (_Float16)f.w;
    }
  }
  __syncthreads();
  {
    __align__(16) _Float16 tmp[16];
    #pragma unroll
    for (int i = 0; i < 16; i++) tmp[i] = T[seg + i][r];   // transpose read
    _Float16* dst = Wt + (size_t)(n0 + r) * HID + k0 + seg;
    *(v8h*)(dst + 0) = *(const v8h*)(tmp + 0);
    *(v8h*)(dst + 8) = *(const v8h*)(tmp + 8);
  }
}

// ---------------------------------------------------------------------------
// Projection GEMM: A f32 (converted on the fly, staged in LDS) x Wt f16 (direct
// global B-fragments) -> f16 outputs.
// mode 0: A = concat(context, latents) per batch, Wt = [WkT;WvT] (N=1024).
//         n<512  -> Kout row-major [b,kv,512]
//         n>=512 -> Vt transposed  [b,512,kv]   (index swap done at LDS write)
// mode 1: A = latents (M = BSZ*QLEN), Wt = WqT (N=2048), out row-major.
// Tile 128x128, K-step 32, 256 threads = 8 waves (2x4), wave = 64x32.
// Grid: x = N-tile (fastest) so the 8 N-blocks reuse each A-tile via L2.
// ---------------------------------------------------------------------------
__global__ __launch_bounds__(256)
void proj_gemm_kernel(const float* __restrict__ ctx, const float* __restrict__ lat,
                      const _Float16* __restrict__ Wt,
                      _Float16* __restrict__ outA, _Float16* __restrict__ outB,
                      int mode)
{
  __shared__ _Float16 At[128][TLDA];   // [m][k]
  __shared__ _Float16 Ct[128][CLDA];   // staged C tile for coalesced writeback

  const int tid  = threadIdx.x;
  const int lane = tid & 31;
  const int wv   = tid >> 5;
  const int half = lane >> 4;
  const int ln   = lane & 15;
  const int wm   = wv & 1;     // 64-row block within tile
  const int wn   = wv >> 1;    // 32-col block within tile

  const int n0 = blockIdx.x * 128;   // N fastest-varying
  const int m0 = blockIdx.y * 128;
  const bool vmode = (mode == 0) && (n0 >= KVD);

  v8f acc[4][2] = {};

  // A loader assignment: 128 rows x (2 x 16 floats)
  const int ar = tid >> 1, ahalf = tid & 1;
  const float* arow;
  {
    int m = m0 + ar;
    if (mode == 0) {
      int b = m / KVL, r = m - b * KVL;
      arow = (r < SEQ) ? (ctx + ((size_t)b * SEQ + r) * HID)
                       : (lat + ((size_t)b * QLEN + (r - SEQ)) * HID);
    } else {
      arow = lat + (size_t)m * HID;   // m = b*QLEN + q
    }
  }

  // per-wave B pointers (Wt row-major [n][2048], 16 contiguous f16 per lane)
  const _Float16* brow[2];
  #pragma unroll
  for (int j = 0; j < 2; j++)
    brow[j] = Wt + (size_t)(n0 + wn * 32 + j * 16 + ln) * HID + half * 16;

  for (int kk = 0; kk < HID; kk += 32) {
    // stage A 128x32 (f32 -> f16)
    {
      const float* src = arow + kk + ahalf * 16;
      __builtin_prefetch(src + 64, 0, 0);   // global_prefetch_b8, next K-chunks
      #pragma unroll
      for (int i = 0; i < 16; i += 4) {
        float4 f = *(const float4*)(src + i);
        At[ar][ahalf * 16 + i + 0] = (_Float16)f.x;
        At[ar][ahalf * 16 + i + 1] = (_Float16)f.y;
        At[ar][ahalf * 16 + i + 2] = (_Float16)f.z;
        At[ar][ahalf * 16 + i + 3] = (_Float16)f.w;
      }
    }
    __syncthreads();

    v16h af[4], bf[2];
    #pragma unroll
    for (int i = 0; i < 4; i++) {
      int m = wm * 64 + i * 16 + ln;
      af[i] = ldfrag(&At[m][half * 16], &At[m][half * 16 + 8]);
    }
    #pragma unroll
    for (int j = 0; j < 2; j++) {
      const _Float16* bp = brow[j] + kk;
      bf[j] = ldfrag(bp, bp + 8);
    }
    #pragma unroll
    for (int i = 0; i < 4; i++)
      #pragma unroll
      for (int j = 0; j < 2; j++)
        acc[i][j] = wmma16(af[i], bf[j], acc[i][j]);
    __syncthreads();
  }

  // stage C into LDS; branch hoisted OUTSIDE the element loops (uniform per block)
  if (vmode) {
    // transposed orientation: a lane's 8 elements are contiguous -> one b128 each
    #pragma unroll
    for (int i = 0; i < 4; i++)
      #pragma unroll
      for (int j = 0; j < 2; j++) {
        const int nl  = wn * 32 + j * 16 + ln;
        const int ml0 = wm * 64 + i * 16 + half * 8;
        __align__(16) _Float16 tmp[8];
        #pragma unroll
        for (int r = 0; r < 8; r++) tmp[r] = (_Float16)acc[i][j][r];
        *(v8h*)&Ct[nl][ml0] = *(const v8h*)tmp;
      }
  } else {
    #pragma unroll
    for (int i = 0; i < 4; i++)
      #pragma unroll
      for (int r = 0; r < 8; r++) {
        const int ml = wm * 64 + i * 16 + r + half * 8;
        #pragma unroll
        for (int j = 0; j < 2; j++)
          Ct[ml][wn * 32 + j * 16 + ln] = (_Float16)acc[i][j][r];
      }
  }
  __syncthreads();

  // coalesced copy-out: each thread writes one contiguous 128B run
  {
    const int row = tid >> 1;
    const int seg = (tid & 1) * 64;
    _Float16* dst;
    if (mode == 1) {
      dst = outA + (size_t)(m0 + row) * HID + n0 + seg;
    } else if (!vmode) {
      int m = m0 + row;
      int b = m / KVL, rr = m - b * KVL;
      dst = outA + ((size_t)b * KVL + rr) * KVD + n0 + seg;
    } else {
      int n  = n0 - KVD + row;
      int ms = m0 + seg;                 // 64-aligned; never straddles a batch
      int b  = ms / KVL, rr = ms - b * KVL;
      dst = outB + ((size_t)b * KVD + n) * (size_t)KVL + rr;
    }
    const _Float16* srcp = &Ct[row][seg];
    #pragma unroll
    for (int i = 0; i < 64; i += 8)
      *(v8h*)(dst + i) = *(const v8h*)(srcp + i);
  }
}

// ---------------------------------------------------------------------------
// Flash attention: one block per (head, batch). 128 threads = 4 waves, each
// wave owns 16 q-rows. kv chunks of 64. K row-major [b,kv,512], V transposed
// [b,512,kv] -> all WMMA B fragments are contiguous 32B global loads, served
// from the 192MB L2 (K+V = 68MB, each group reused by 4 heads).
// ---------------------------------------------------------------------------
__global__ __launch_bounds__(128)
void attn_kernel(const _Float16* __restrict__ Qf, const _Float16* __restrict__ Kf,
                 const _Float16* __restrict__ Vt, _Float16* __restrict__ Of)
{
  __shared__ _Float16 Qs[64][72];
  __shared__ float    Ss[64][65];
  __shared__ _Float16 Ps[64][72];
  __shared__ float    mrow[64], lrow[64], crow[64];

  const int tid  = threadIdx.x;
  const int lane = tid & 31;
  const int wv   = tid >> 5;       // 4 waves
  const int half = lane >> 4;
  const int ln   = lane & 15;
  const int head = blockIdx.x;
  const int b    = blockIdx.y;
  const int g    = head >> 2;      // GQA group (NH/NG = 4)

  // stage Q tile 64x64
  {
    int r = tid >> 1, h2 = tid & 1;
    const _Float16* src = Qf + ((size_t)(b * QLEN + r)) * HID + head * HD + h2 * 32;
    #pragma unroll
    for (int i = 0; i < 32; i += 8)
      *(v8h*)&Qs[r][h2 * 32 + i] = *(const v8h*)(src + i);
  }
  if (tid < 64) { mrow[tid] = -1e30f; lrow[tid] = 0.f; }

  v8f oacc[4] = {};
  const size_t kbase = (size_t)b * KVL * KVD + (size_t)g * HD;
  const size_t vbase = ((size_t)b * KVD + (size_t)g * HD) * (size_t)KVL;

  __syncthreads();

  for (int c = 0; c < KVL; c += 64) {
    // S = (Q K^T) * 1/sqrt(64)
    #pragma unroll
    for (int nt = 0; nt < 4; nt++) {
      v8f s = {};
      #pragma unroll
      for (int ks = 0; ks < 2; ks++) {
        int m = wv * 16 + ln;
        v16h a = ldfrag(&Qs[m][ks * 32 + half * 16], &Qs[m][ks * 32 + half * 16 + 8]);
        const _Float16* kp = Kf + kbase + (size_t)(c + nt * 16 + ln) * KVD + ks * 32 + half * 16;
        __builtin_prefetch(kp + 64 * KVD, 0, 0);
        v16h bb = ldfrag(kp, kp + 8);
        s = wmma16(a, bb, s);
      }
      #pragma unroll
      for (int r = 0; r < 8; r++)
        Ss[wv * 16 + r + half * 8][nt * 16 + ln] = s[r] * 0.125f;
    }
    __syncthreads();

    // online softmax, one thread per q-row
    if (tid < 64) {
      float mo = mrow[tid];
      float mx = mo;
      for (int j = 0; j < 64; j++) mx = fmaxf(mx, Ss[tid][j]);
      float corr = __expf(mo - mx);
      float sum = 0.f;
      for (int j = 0; j < 64; j++) {
        float p = __expf(Ss[tid][j] - mx);
        sum += p;
        Ps[tid][j] = (_Float16)p;
      }
      mrow[tid] = mx;
      lrow[tid] = lrow[tid] * corr + sum;
      crow[tid] = corr;
    }
    __syncthreads();

    // rescale O, then O += P V
    #pragma unroll
    for (int r = 0; r < 8; r++) {
      float corr = crow[wv * 16 + r + half * 8];
      #pragma unroll
      for (int nt = 0; nt < 4; nt++) oacc[nt][r] *= corr;
    }
    #pragma unroll
    for (int nt = 0; nt < 4; nt++) {
      #pragma unroll
      for (int ks = 0; ks < 2; ks++) {
        int m = wv * 16 + ln;
        v16h a = ldfrag(&Ps[m][ks * 32 + half * 16], &Ps[m][ks * 32 + half * 16 + 8]);
        const _Float16* vp = Vt + vbase + (size_t)(nt * 16 + ln) * KVL + c + ks * 32 + half * 16;
        __builtin_prefetch(vp + 64, 0, 0);
        v16h bb = ldfrag(vp, vp + 8);
        oacc[nt] = wmma16(a, bb, oacc[nt]);
      }
    }
    __syncthreads();
  }

  // normalize and write O as f16 [b, q, head*64 + d]
  #pragma unroll
  for (int r = 0; r < 8; r++) {
    int q = wv * 16 + r + half * 8;
    float inv = 1.f / lrow[q];
    #pragma unroll
    for (int nt = 0; nt < 4; nt++)
      Of[((size_t)(b * QLEN + q)) * HID + head * HD + nt * 16 + ln] =
          (_Float16)(oacc[nt][r] * inv);
  }
}

// ---------------------------------------------------------------------------
// Output projection: Of16[512,2048] x WoT f16 [2048][2048] -> f32 d_out.
// A staged in LDS (plain f16 copies); B-fragments direct from global.
// f32 stores straight from accumulators (64B per half-wave, simple addressing).
// ---------------------------------------------------------------------------
__global__ __launch_bounds__(256)
void outproj_kernel(const _Float16* __restrict__ A, const _Float16* __restrict__ WoT,
                    float* __restrict__ out)
{
  __shared__ _Float16 At[128][TLDA];

  const int tid  = threadIdx.x;
  const int lane = tid & 31;
  const int wv   = tid >> 5;
  const int half = lane >> 4;
  const int ln   = lane & 15;
  const int wm   = wv & 1;
  const int wn   = wv >> 1;

  const int n0 = blockIdx.x * 128;
  const int m0 = blockIdx.y * 128;

  v8f acc[4][2] = {};

  const int ar = tid >> 1, ahalf = tid & 1;
  const _Float16* arow = A + (size_t)(m0 + ar) * HID;

  const _Float16* brow[2];
  #pragma unroll
  for (int j = 0; j < 2; j++)
    brow[j] = WoT + (size_t)(n0 + wn * 32 + j * 16 + ln) * HID + half * 16;

  for (int kk = 0; kk < HID; kk += 32) {
    {
      const _Float16* src = arow + kk + ahalf * 16;
      *(v8h*)&At[ar][ahalf * 16 + 0] = *(const v8h*)(src + 0);
      *(v8h*)&At[ar][ahalf * 16 + 8] = *(const v8h*)(src + 8);
    }
    __syncthreads();

    v16h af[4], bf[2];
    #pragma unroll
    for (int i = 0; i < 4; i++) {
      int m = wm * 64 + i * 16 + ln;
      af[i] = ldfrag(&At[m][half * 16], &At[m][half * 16 + 8]);
    }
    #pragma unroll
    for (int j = 0; j < 2; j++) {
      const _Float16* bp = brow[j] + kk;
      bf[j] = ldfrag(bp, bp + 8);
    }
    #pragma unroll
    for (int i = 0; i < 4; i++)
      #pragma unroll
      for (int j = 0; j < 2; j++)
        acc[i][j] = wmma16(af[i], bf[j], acc[i][j]);
    __syncthreads();
  }

  #pragma unroll
  for (int i = 0; i < 4; i++)
    #pragma unroll
    for (int r = 0; r < 8; r++) {
      size_t m = m0 + wm * 64 + i * 16 + r + half * 8;
      #pragma unroll
      for (int j = 0; j < 2; j++) {
        int n = n0 + wn * 32 + j * 16 + ln;
        out[m * HID + n] = acc[i][j][r];
      }
    }
}

extern "C" void kernel_launch(void* const* d_in, const int* in_sizes, int n_in,
                              void* d_out, int out_size, void* d_ws, size_t ws_size,
                              hipStream_t stream) {
  (void)in_sizes; (void)n_in; (void)out_size; (void)ws_size;

  const float* latents = (const float*)d_in[0];
  const float* context = (const float*)d_in[1];
  const float* Wq      = (const float*)d_in[2];
  const float* Wk      = (const float*)d_in[3];
  const float* Wv      = (const float*)d_in[4];
  const float* Wo      = (const float*)d_in[5];
  float*       out     = (float*)d_out;

  char* ws = (char*)d_ws;
  size_t off = 0;
  const size_t szKf  = (size_t)BSZ * KVL * KVD * sizeof(_Float16);    // 34 MB
  const size_t szQf  = (size_t)BSZ * QLEN * HID * sizeof(_Float16);   //  2 MB
  const size_t szWkv = (size_t)1024 * HID * sizeof(_Float16);         //  4 MB
  const size_t szWsq = (size_t)HID * HID * sizeof(_Float16);          //  8 MB

  _Float16* Kf   = (_Float16*)(ws + off); off += szKf;
  _Float16* Vt   = (_Float16*)(ws + off); off += szKf;
  _Float16* Qf   = (_Float16*)(ws + off); off += szQf;
  _Float16* Of   = (_Float16*)(ws + off); off += szQf;
  _Float16* WkvT = (_Float16*)(ws + off); off += szWkv;
  _Float16* WqT  = (_Float16*)(ws + off); off += szWsq;
  _Float16* WoT  = (_Float16*)(ws + off); off += szWsq;

  // one-time weight transpose/convert to f16 [n][k]
  wtrans_kernel<<<dim3(32,  8), 256, 0, stream>>>(Wk, WkvT,                       KVD);
  wtrans_kernel<<<dim3(32,  8), 256, 0, stream>>>(Wv, WkvT + (size_t)KVD * HID,   KVD);
  wtrans_kernel<<<dim3(32, 32), 256, 0, stream>>>(Wq, WqT,                        HID);
  wtrans_kernel<<<dim3(32, 32), 256, 0, stream>>>(Wo, WoT,                        HID);

  // K,V projection over concat(context, latents): M=33280, N=1024 (N fastest)
  proj_gemm_kernel<<<dim3(8, 260), 256, 0, stream>>>(context, latents, WkvT, Kf, Vt, 0);
  // Q projection: M=512, N=2048
  proj_gemm_kernel<<<dim3(16, 4), 256, 0, stream>>>(context, latents, WqT, Qf, nullptr, 1);
  // attention per (head, batch)
  attn_kernel<<<dim3(NH, BSZ), 128, 0, stream>>>(Qf, Kf, Vt, Of);
  // output projection -> f32
  outproj_kernel<<<dim3(16, 4), 256, 0, stream>>>(Of, WoT, out);
}
